// PointTransformerSeg_15204184227909
// MI455X (gfx1250) — matrix operand order repaired and checked by reading, not compiled, not run on hardware
//
#include <hip/hip_runtime.h>
#include <hip/hip_bf16.h>
#include <math.h>

typedef __attribute__((ext_vector_type(16))) _Float16 v16h;
typedef __attribute__((ext_vector_type(8)))  _Float16 v8h;
typedef __attribute__((ext_vector_type(8)))  float    v8f;

#define CDIV(a, b) (((a) + (b) - 1) / (b))

// ---------------------------------------------------------------------------
// WMMA GEMM: C[M,N] = act(A[M,K] @ W[K,N] + bias)
// f32 storage, f16 multiply via v_wmma_f32_16x16x32_f16, f32 accumulate.
// Block = 256 threads = 8 waves. Block tile 64(M) x 64(N), K-step 32.
// Waves arranged 4(M) x 2(N); each wave owns a 16x32 output strip => 2 WMMAs.
//
// Double-buffered LDS pipeline: while tile i feeds the WMMAs, tile i+1's
// global loads are in flight; one barrier per K-step. Staging is branch-free
// (clamped in-bounds addresses + select-zero), A row bases hoisted out of the
// loop, B row offsets carried as running values (no muls in the loop).
// Fragments come out of LDS as 16B ds_load_b128 chunks (rows padded +8
// halves -> 20/36-bank strides, conflict-free).
//   A (16-bit 16x32, ISA 7.12.2): lane L holds row M=L&15;
//     halves e=0..7  -> K = grp*8 + e        (chunk at lA[m][grp*8])
//     halves e=8..15 -> K = 16 + grp*8 + e-8 (chunk at lA[m][16+grp*8])
//   B (16-bit 32x16): lane L holds N=L&15; halves e -> K same as A (column gather).
// ---------------------------------------------------------------------------
#define BM 64
#define BN 64
#define BK 32

__global__ void gemm_wmma_f16_kernel(const float* __restrict__ A,
                                     const float* __restrict__ W,
                                     const float* __restrict__ bias,
                                     float* __restrict__ C,
                                     int M, int K, int N, int relu) {
  __builtin_amdgcn_s_wait_tensorcnt((short)0);   // CDNA5 split-counter path
  __shared__ _Float16 lA[2][BM][BK + 8];         // +8 halves: bank destagger
  __shared__ _Float16 lB[2][BK][BN + 8];

  const int tid  = threadIdx.x;
  const int lane = tid & 31;
  const int wave = tid >> 5;          // 0..7
  const int wm   = wave & 3;          // 4 waves along M
  const int wn   = wave >> 2;         // 2 waves along N
  const int tm0  = blockIdx.y * BM;
  const int tn0  = blockIdx.x * BN;
  const int grp  = lane >> 4;         // K-half select
  const int mloc = wm * 16 + (lane & 15);

  // ---- fixed per-thread staging coordinates ----
  int    rA[4], kA[4];  bool aOK[4];  size_t rowA[4];
  int    rBr[4], nB[4]; bool n0OK[4], n1OK[4]; size_t rowB[4];
#pragma unroll
  for (int i = 0; i < 4; ++i) {
    int t = tid + i * 256;
    rA[i] = t >> 4; kA[i] = (t & 15) << 1;
    int gm = tm0 + rA[i];
    aOK[i]  = (gm < M);
    rowA[i] = aOK[i] ? (size_t)gm * K : (size_t)0;
    rBr[i] = t >> 5; nB[i] = (t & 31) << 1;
    int gn = tn0 + nB[i];
    n0OK[i] = (gn < N);
    n1OK[i] = (gn + 1 < N);
    rowB[i] = (size_t)rBr[i] * N;     // advances by BK*N per staged tile
  }

  float va[4][2], vb[4][2];

  // branch-free global fetch of tile at K-offset k0 (advances rowB)
  auto stage_load = [&](int k0) {
#pragma unroll
    for (int i = 0; i < 4; ++i) {
      int gk = k0 + kA[i];
      bool v0 = aOK[i] && (gk < K);
      bool v1 = aOK[i] && (gk + 1 < K);
      va[i][0] = A[rowA[i] + (size_t)(v0 ? gk : 0)];
      va[i][1] = A[rowA[i] + (size_t)(v1 ? gk + 1 : 0)];
    }
#pragma unroll
    for (int i = 0; i < 4; ++i) {
      int gk = k0 + rBr[i];
      bool kOK = (gk < K);
      bool v0 = kOK && n0OK[i];
      bool v1 = kOK && n1OK[i];
      int gn = tn0 + nB[i];
      vb[i][0] = W[v0 ? (rowB[i] + (size_t)gn) : (size_t)0];
      vb[i][1] = W[v1 ? (rowB[i] + (size_t)gn + 1) : (size_t)0];
      rowB[i] += (size_t)BK * N;
    }
  };
  // convert + packed f16-pair stores into LDS buffer `buf`
  auto stage_store = [&](int buf, int k0) {
#pragma unroll
    for (int i = 0; i < 4; ++i) {
      int gk = k0 + kA[i];
      bool v0 = aOK[i] && (gk < K);
      bool v1 = aOK[i] && (gk + 1 < K);
      union { _Float16 h[2]; unsigned u; } p;
      p.h[0] = (_Float16)(v0 ? va[i][0] : 0.0f);
      p.h[1] = (_Float16)(v1 ? va[i][1] : 0.0f);
      *(unsigned*)&lA[buf][rA[i]][kA[i]] = p.u;
    }
#pragma unroll
    for (int i = 0; i < 4; ++i) {
      int gk = k0 + rBr[i];
      bool kOK = (gk < K);
      bool v0 = kOK && n0OK[i];
      bool v1 = kOK && n1OK[i];
      union { _Float16 h[2]; unsigned u; } p;
      p.h[0] = (_Float16)(v0 ? vb[i][0] : 0.0f);
      p.h[1] = (_Float16)(v1 ? vb[i][1] : 0.0f);
      *(unsigned*)&lB[buf][rBr[i]][nB[i]] = p.u;
    }
  };

  v8f acc0 = {};
  v8f acc1 = {};

  // ---- pipeline prologue: stage tile 0 ----
  stage_load(0);
  stage_store(0, 0);
  __syncthreads();

  const int nk = CDIV(K, BK);
  int buf = 0;
  for (int ik = 0; ik < nk; ++ik) {
    const int k0 = ik * BK;
    const bool has_next = (ik + 1 < nk);

    // ---- fragment fetch from LDS (ds_load_b128 chunks) ----
    v16h a;
    {
      v8h lo = *(const v8h*)&lA[buf][mloc][grp * 8];
      v8h hi = *(const v8h*)&lA[buf][mloc][16 + grp * 8];
#pragma unroll
      for (int e = 0; e < 8; ++e) { a[e] = lo[e]; a[8 + e] = hi[e]; }
    }
    v16h b0, b1;
    {
      const int nl = wn * 32 + (lane & 15);
#pragma unroll
      for (int e = 0; e < 8; ++e) {
        b0[e] = lB[buf][grp * 8 + e][nl]; b0[8 + e] = lB[buf][16 + grp * 8 + e][nl];
        b1[e] = lB[buf][grp * 8 + e][nl + 16]; b1[8 + e] = lB[buf][16 + grp * 8 + e][nl + 16];
      }
    }

    // ---- issue next tile's global loads; latency hides under WMMAs ----
    if (has_next) {
      if (k0 + 2 * BK < K) {          // keep prefetch two slabs ahead
        int pr = tm0 + (tid & (BM - 1));
        pr = pr < M ? pr : (M - 1);
        __builtin_prefetch(A + (size_t)pr * K + k0 + 2 * BK, 0, 3);
      }
      stage_load(k0 + BK);
    }

    acc0 = __builtin_amdgcn_wmma_f32_16x16x32_f16(false, a, false, b0,
                                                  (short)0, acc0, false, false);
    acc1 = __builtin_amdgcn_wmma_f32_16x16x32_f16(false, a, false, b1,
                                                  (short)0, acc1, false, false);

    if (has_next) stage_store(buf ^ 1, k0 + BK);
    __syncthreads();
    buf ^= 1;
  }

  // ---- store: C layout: VGPR r -> M = r + 8*(lane>=16), N = lane&15 ----
  const int gm_base = tm0 + wm * 16;
  const int gn0 = tn0 + wn * 32 + (lane & 15);
  const int gn1 = gn0 + 16;
#pragma unroll
  for (int r = 0; r < 8; ++r) {
    int m = gm_base + r + grp * 8;
    if (m < M) {
      if (gn0 < N) {
        float v = acc0[r];
        if (bias) v += bias[gn0];
        if (relu) v = fmaxf(v, 0.0f);
        C[(size_t)m * N + gn0] = v;
      }
      if (gn1 < N) {
        float v = acc1[r];
        if (bias) v += bias[gn1];
        if (relu) v = fmaxf(v, 0.0f);
        C[(size_t)m * N + gn1] = v;
      }
    }
  }
}

// ---------------------------------------------------------------------------
// BatchNorm (training-mode biased batch stats), two-pass.
// ---------------------------------------------------------------------------
__global__ void bn_stats_kernel(const float* __restrict__ X, int rows, int C,
                                float* __restrict__ mean, float* __restrict__ var) {
  const int c = blockIdx.x;
  float s = 0.0f, s2 = 0.0f;
  for (int r = threadIdx.x; r < rows; r += blockDim.x) {
    float v = X[(size_t)r * C + c];
    s += v; s2 += v * v;
  }
  __shared__ float sh[256];
  __shared__ float sh2[256];
  sh[threadIdx.x] = s; sh2[threadIdx.x] = s2;
  __syncthreads();
  for (int st = 128; st > 0; st >>= 1) {
    if ((int)threadIdx.x < st) {
      sh[threadIdx.x] += sh[threadIdx.x + st];
      sh2[threadIdx.x] += sh2[threadIdx.x + st];
    }
    __syncthreads();
  }
  if (threadIdx.x == 0) {
    float m = sh[0] / (float)rows;
    mean[c] = m;
    var[c] = sh2[0] / (float)rows - m * m;
  }
}

__global__ void bn_apply_kernel(float* __restrict__ X, size_t total, int C,
                                const float* __restrict__ mean,
                                const float* __restrict__ var,
                                const float* __restrict__ g,
                                const float* __restrict__ b, int relu) {
  size_t t = (size_t)blockIdx.x * blockDim.x + threadIdx.x;
  if (t >= total) return;
  int c = (int)(t % C);
  float v = (X[t] - mean[c]) * rsqrtf(var[c] + 1e-5f) * g[c] + b[c];
  if (relu) v = fmaxf(v, 0.0f);
  X[t] = v;
}

// ---------------------------------------------------------------------------
// Brute-force KNN: per-lane insertion-sorted top-k (k <= 16).
// ---------------------------------------------------------------------------
__global__ void knn_kernel(const float* __restrict__ Q, int nq,
                           const float* __restrict__ S, int nsrc,
                           int k, int* __restrict__ idx_out) {
  int i = blockIdx.x * blockDim.x + threadIdx.x;
  if (i >= nq) return;
  float qx = Q[i * 3 + 0], qy = Q[i * 3 + 1], qz = Q[i * 3 + 2];
  float bd[16]; int bi[16];
  for (int t = 0; t < k; ++t) { bd[t] = 3.0e38f; bi[t] = 0; }
  for (int j = 0; j < nsrc; ++j) {
    float dx = S[j * 3 + 0] - qx;
    float dy = S[j * 3 + 1] - qy;
    float dz = S[j * 3 + 2] - qz;
    float d = dx * dx + dy * dy + dz * dz;
    if (d < bd[k - 1]) {
      int at = k - 1;
      while (at > 0 && bd[at - 1] > d) {
        bd[at] = bd[at - 1]; bi[at] = bi[at - 1]; --at;
      }
      bd[at] = d; bi[at] = j;
    }
  }
  for (int t = 0; t < k; ++t) idx_out[(size_t)i * k + t] = bi[t];
}

// ---------------------------------------------------------------------------
// Furthest-point sampling, seeded at 0: one 1024-thread workgroup (32 waves),
// LDS tree argmax per iteration.
// ---------------------------------------------------------------------------
__global__ void fps_kernel(const float* __restrict__ P, int n, int m,
                           int* __restrict__ idx, float* __restrict__ dists) {
  const int tid = threadIdx.x;
  const int nt = blockDim.x;
  __shared__ float sd[1024];
  __shared__ int si[1024];
  __shared__ int s_last;
  for (int i = tid; i < n; i += nt) dists[i] = 1e10f;
  if (tid == 0) { idx[0] = 0; s_last = 0; }
  __syncthreads();
  int last = 0;
  for (int it = 1; it < m; ++it) {
    float lx = P[last * 3 + 0], ly = P[last * 3 + 1], lz = P[last * 3 + 2];
    float bd = -1.0f; int bi = 0;
    for (int j = tid; j < n; j += nt) {
      float dx = P[j * 3 + 0] - lx;
      float dy = P[j * 3 + 1] - ly;
      float dz = P[j * 3 + 2] - lz;
      float d = dx * dx + dy * dy + dz * dz;
      float nd = fminf(dists[j], d);
      dists[j] = nd;
      if (nd > bd) { bd = nd; bi = j; }
    }
    sd[tid] = bd; si[tid] = bi;
    __syncthreads();
    for (int st = nt >> 1; st > 0; st >>= 1) {
      if (tid < st && sd[tid + st] > sd[tid]) {
        sd[tid] = sd[tid + st]; si[tid] = si[tid + st];
      }
      __syncthreads();
    }
    if (tid == 0) { idx[it] = si[0]; s_last = si[0]; }
    __syncthreads();
    last = s_last;
    __syncthreads();
  }
}

// ---------------------------------------------------------------------------
// Gathers / grouping / elementwise helpers
// ---------------------------------------------------------------------------
__global__ void gather_rows_kernel(const float* __restrict__ S,
                                   const int* __restrict__ idx,
                                   int n, int c, float* __restrict__ out) {
  size_t t = (size_t)blockIdx.x * blockDim.x + threadIdx.x;
  if (t >= (size_t)n * c) return;
  int i = (int)(t / c), cc = (int)(t % c);
  out[t] = S[(size_t)idx[i] * c + cc];
}

// G[m, ns, 3+c] = [ P[idx]-NP[i], X[idx] ]   (c==0 -> coords only)
__global__ void group_cat_kernel(const float* __restrict__ P,
                                 const float* __restrict__ X,
                                 const float* __restrict__ NP,
                                 const int* __restrict__ nidx,
                                 int m, int ns, int c, float* __restrict__ G) {
  int t = blockIdx.x * blockDim.x + threadIdx.x;
  if (t >= m * ns) return;
  int i = t / ns;
  int src = nidx[t];
  float* g = G + (size_t)t * (3 + c);
  g[0] = P[src * 3 + 0] - NP[i * 3 + 0];
  g[1] = P[src * 3 + 1] - NP[i * 3 + 1];
  g[2] = P[src * 3 + 2] - NP[i * 3 + 2];
  for (int cc = 0; cc < c; ++cc) g[3 + cc] = X[(size_t)src * c + cc];
}

__global__ void max_over_ns_kernel(const float* __restrict__ H, int m, int ns,
                                   int c, float* __restrict__ out) {
  size_t t = (size_t)blockIdx.x * blockDim.x + threadIdx.x;
  if (t >= (size_t)m * c) return;
  int i = (int)(t / c), cc = (int)(t % c);
  float mx = -3.0e38f;
  for (int j = 0; j < ns; ++j)
    mx = fmaxf(mx, H[((size_t)i * ns + j) * c + cc]);
  out[t] = mx;
}

// w0 = K[idx] - Q[i] + PR
__global__ void att_w0_kernel(const float* __restrict__ Kf,
                              const float* __restrict__ Qf,
                              const float* __restrict__ PR,
                              const int* __restrict__ idx,
                              int n, int ns, int c, float* __restrict__ W0) {
  size_t t = (size_t)blockIdx.x * blockDim.x + threadIdx.x;
  if (t >= (size_t)n * ns) return;
  int i = (int)(t / ns);
  int src = idx[t];
  for (int cc = 0; cc < c; ++cc)
    W0[t * c + cc] = Kf[(size_t)src * c + cc] - Qf[(size_t)i * c + cc] + PR[t * c + cc];
}

// xv = V[idx] + PR
__global__ void att_xv_kernel(const float* __restrict__ V,
                              const float* __restrict__ PR,
                              const int* __restrict__ idx,
                              int n, int ns, int c, float* __restrict__ XV) {
  size_t t = (size_t)blockIdx.x * blockDim.x + threadIdx.x;
  if (t >= (size_t)n * ns) return;
  int src = idx[t];
  for (int cc = 0; cc < c; ++cc)
    XV[t * c + cc] = V[(size_t)src * c + cc] + PR[t * c + cc];
}

// softmax over ns axis, per (point, shared-channel)
__global__ void softmax_ns_kernel(float* __restrict__ W, int n, int ns, int cs) {
  size_t t = (size_t)blockIdx.x * blockDim.x + threadIdx.x;
  if (t >= (size_t)n * cs) return;
  int i = (int)(t / cs), ch = (int)(t % cs);
  float mx = -3.0e38f;
  for (int j = 0; j < ns; ++j)
    mx = fmaxf(mx, W[((size_t)i * ns + j) * cs + ch]);
  float s = 0.0f;
  for (int j = 0; j < ns; ++j) {
    size_t o = ((size_t)i * ns + j) * cs + ch;
    float e = expf(W[o] - mx);
    W[o] = e; s += e;
  }
  float inv = 1.0f / s;
  for (int j = 0; j < ns; ++j)
    W[((size_t)i * ns + j) * cs + ch] *= inv;
}

// out[i,cc] = sum_j XV[i,j,cc] * W[i,j,cc%cs]
__global__ void att_agg_kernel(const float* __restrict__ XV,
                               const float* __restrict__ W,
                               int n, int ns, int c, int cs,
                               float* __restrict__ out) {
  size_t t = (size_t)blockIdx.x * blockDim.x + threadIdx.x;
  if (t >= (size_t)n * c) return;
  int i = (int)(t / c), cc = (int)(t % c);
  int ch = cc % cs;
  float acc = 0.0f;
  for (int j = 0; j < ns; ++j)
    acc += XV[((size_t)i * ns + j) * c + cc] * W[((size_t)i * ns + j) * cs + ch];
  out[t] = acc;
}

__global__ void mean_rows_kernel(const float* __restrict__ X, int n, int c,
                                 float* __restrict__ out) {
  int ch = blockIdx.x * blockDim.x + threadIdx.x;
  if (ch >= c) return;
  float s = 0.0f;
  for (int r = 0; r < n; ++r) s += X[(size_t)r * c + ch];
  out[ch] = s / (float)n;
}

__global__ void concat2_kernel(const float* __restrict__ A, int ca,
                               const float* __restrict__ B, int cb,
                               int n, int bcastB, float* __restrict__ out) {
  size_t t = (size_t)blockIdx.x * blockDim.x + threadIdx.x;
  size_t total = (size_t)n * (ca + cb);
  if (t >= total) return;
  int row = (int)(t / (ca + cb));
  int ch = (int)(t % (ca + cb));
  out[t] = (ch < ca) ? A[(size_t)row * ca + ch]
                     : B[(size_t)(bcastB ? 0 : row) * cb + (ch - ca)];
}

// A[i,:] += sum_k wgt_k * F2[idx3[i,k], :], wgt = normalized 1/(dist+eps)
__global__ void interp3_kernel(const float* __restrict__ P1,
                               const float* __restrict__ P2,
                               const int* __restrict__ idx3,
                               const float* __restrict__ F2,
                               float* __restrict__ A, int n1, int c) {
  int i = blockIdx.x * blockDim.x + threadIdx.x;
  if (i >= n1) return;
  float x = P1[i * 3 + 0], y = P1[i * 3 + 1], z = P1[i * 3 + 2];
  float w[3]; int jj[3]; float s = 0.0f;
  for (int k = 0; k < 3; ++k) {
    int j = idx3[i * 3 + k];
    jj[k] = j;
    float dx = x - P2[j * 3 + 0];
    float dy = y - P2[j * 3 + 1];
    float dz = z - P2[j * 3 + 2];
    float r = 1.0f / (sqrtf(dx * dx + dy * dy + dz * dz) + 1e-8f);
    w[k] = r; s += r;
  }
  float inv = 1.0f / s;
  for (int k = 0; k < 3; ++k) w[k] *= inv;
  for (int cc = 0; cc < c; ++cc) {
    float acc = 0.0f;
    for (int k = 0; k < 3; ++k) acc += w[k] * F2[(size_t)jj[k] * c + cc];
    A[(size_t)i * c + cc] += acc;
  }
}

__global__ void add_relu_kernel(const float* __restrict__ H,
                                float* __restrict__ X, size_t total) {
  size_t t = (size_t)blockIdx.x * blockDim.x + threadIdx.x;
  if (t >= total) return;
  X[t] = fmaxf(H[t] + X[t], 0.0f);
}

// ---------------------------------------------------------------------------
// Host orchestration
// ---------------------------------------------------------------------------
struct BNp  { const float* b; const float* g; };
struct Linp { const float* b; const float* w; };
struct Blkp {
  BNp bn1, bn2, bn3;
  const float* lin1; const float* lin3;
  Linp k, p1, p2; BNp pbn; Linp q, v, w1, w2; BNp wbn1, wbn2;
};
struct TUp { BNp bn1, bn2; Linp l1, l2; };
struct TDp { const float* b; const float* g; const float* w; };

extern "C" void kernel_launch(void* const* d_in, const int* in_sizes, int n_in,
                              void* d_out, int out_size, void* d_ws, size_t ws_size,
                              hipStream_t stream) {
  (void)out_size; (void)ws_size;
  const float* pos  = (const float*)d_in[0];
  const float* feat = (const float*)d_in[1];
  const int n1 = in_sizes[0] / 3;            // 16384
  const int n2 = n1 / 4, n3 = n2 / 4, n4 = n3 / 4, n5 = n4 / 4;
  const int NS[5] = {8, 16, 16, 16, 16};

  // ---- parse params in JAX pytree (sorted-key DFS) order -------------------
  int cur = 2;
  auto nextp = [&]() -> const float* {
    int i = (cur < n_in) ? cur : (n_in - 1);
    ++cur;
    return (const float*)d_in[i];
  };
  auto parseBN  = [&]() { BNp b;  b.b = nextp(); b.g = nextp(); return b; };
  auto parseLin = [&]() { Linp l; l.b = nextp(); l.w = nextp(); return l; };
  auto parseBlk = [&]() {
    Blkp b;
    b.bn1 = parseBN(); b.bn2 = parseBN(); b.bn3 = parseBN();
    b.lin1 = nextp(); b.lin3 = nextp();
    b.k = parseLin(); b.p1 = parseLin(); b.p2 = parseLin();
    b.pbn = parseBN();
    b.q = parseLin(); b.v = parseLin();
    b.w1 = parseLin(); b.w2 = parseLin();
    b.wbn1 = parseBN(); b.wbn2 = parseBN();
    return b;
  };
  auto parseTU = [&](bool head) {
    TUp t;
    t.bn1 = parseBN();
    if (!head) t.bn2 = parseBN();
    t.l1 = parseLin(); t.l2 = parseLin();
    return t;
  };
  BNp  cls_bn = parseBN();
  Linp cls_l1 = parseLin();
  Linp cls_l2 = parseLin();
  Blkp blkd[5]; TUp tud[5];
  for (int i = 0; i < 4; ++i) { blkd[i] = parseBlk(); tud[i] = parseTU(false); }
  blkd[4] = parseBlk(); tud[4] = parseTU(true);
  TDp enc[5];
  for (int i = 0; i < 5; ++i) { enc[i].b = nextp(); enc[i].g = nextp(); enc[i].w = nextp(); }

  // ---- workspace bump allocator --------------------------------------------
  size_t off = 0;
  auto alloc = [&](size_t bytes) -> void* {
    void* p = (void*)((char*)d_ws + off);
    off += (bytes + 255) & ~(size_t)255;
    return p;
  };
  auto fAlloc = [&](size_t elems) { return (float*)alloc(elems * sizeof(float)); };

  // ---- launch helpers ------------------------------------------------------
  auto linw = [&](const float* A, const float* W, const float* bias, float* C,
                  int M, int K, int N, int relu) {
    dim3 grid(CDIV(N, BN), CDIV(M, BM));
    gemm_wmma_f16_kernel<<<grid, 256, 0, stream>>>(A, W, bias, C, M, K, N, relu);
  };
  auto linear = [&](const float* A, const Linp& L, float* C, int M, int K, int N,
                    int relu) { linw(A, L.w, L.b, C, M, K, N, relu); };
  auto bnrelu = [&](float* X, int rows, int C, const BNp& bn, int relu) {
    float* st = fAlloc(2 * (size_t)C);
    bn_stats_kernel<<<C, 256, 0, stream>>>(X, rows, C, st, st + C);
    size_t total = (size_t)rows * C;
    bn_apply_kernel<<<CDIV(total, 256), 256, 0, stream>>>(X, total, C, st, st + C,
                                                          bn.g, bn.b, relu);
  };

  // ---- persistent buffers --------------------------------------------------
  float* x0 = fAlloc((size_t)n1 * 6);
  float* x1 = fAlloc((size_t)n1 * 32);
  float* x2 = fAlloc((size_t)n2 * 64);
  float* x3 = fAlloc((size_t)n3 * 128);
  float* x4 = fAlloc((size_t)n4 * 256);
  float* x5 = fAlloc((size_t)n5 * 512);
  float* p2 = fAlloc((size_t)n2 * 3);
  float* p3 = fAlloc((size_t)n3 * 3);
  float* p4 = fAlloc((size_t)n4 * 3);
  float* p5 = fAlloc((size_t)n5 * 3);
  float* y1 = fAlloc((size_t)n1 * 32);
  float* y2 = fAlloc((size_t)n2 * 64);
  float* y3 = fAlloc((size_t)n3 * 128);
  float* y4 = fAlloc((size_t)n4 * 256);
  float* y5 = fAlloc((size_t)n5 * 512);
  const float* p1 = pos;

  // ---- composite stages ----------------------------------------------------
  auto td = [&](const float* P, const float* X, int n, int cin, int ns,
                const TDp& E, int cout, float* Pout, float* Xout) {
    size_t mark = off;
    int m = n / 4;
    float* dists = fAlloc((size_t)n);
    int* fidx = (int*)alloc((size_t)m * sizeof(int));
    fps_kernel<<<1, 1024, 0, stream>>>(P, n, m, fidx, dists);
    gather_rows_kernel<<<CDIV((size_t)m * 3, 256), 256, 0, stream>>>(P, fidx, m, 3, Pout);
    int* nidx = (int*)alloc((size_t)m * ns * sizeof(int));
    knn_kernel<<<CDIV(m, 64), 64, 0, stream>>>(Pout, m, P, n, ns, nidx);
    float* g = fAlloc((size_t)m * ns * (3 + cin));
    group_cat_kernel<<<CDIV(m * ns, 256), 256, 0, stream>>>(P, X, Pout, nidx, m, ns, cin, g);
    float* h = fAlloc((size_t)m * ns * cout);
    linw(g, E.w, nullptr, h, m * ns, 3 + cin, cout, 0);
    BNp bn = {E.b, E.g};
    bnrelu(h, m * ns, cout, bn, 1);
    max_over_ns_kernel<<<CDIV((size_t)m * cout, 256), 256, 0, stream>>>(h, m, ns, cout, Xout);
    off = mark;
  };

  auto pt = [&](const float* P, const float* X, const Blkp& B, int n, int c,
                int ns, float* out) {
    size_t mark = off;
    int cs = c / 8;
    float* q = fAlloc((size_t)n * c);
    float* kk = fAlloc((size_t)n * c);
    float* v = fAlloc((size_t)n * c);
    linear(X, B.q, q, n, c, c, 0);
    linear(X, B.k, kk, n, c, c, 0);
    linear(X, B.v, v, n, c, c, 0);
    int* idx = (int*)alloc((size_t)n * ns * sizeof(int));
    knn_kernel<<<CDIV(n, 64), 64, 0, stream>>>(P, n, P, n, ns, idx);
    float* pr0 = fAlloc((size_t)n * ns * 3);
    group_cat_kernel<<<CDIV(n * ns, 256), 256, 0, stream>>>(P, (const float*)nullptr, P,
                                                            idx, n, ns, 0, pr0);
    float* t1 = fAlloc((size_t)n * ns * 3);
    linear(pr0, B.p1, t1, n * ns, 3, 3, 0);
    bnrelu(t1, n * ns, 3, B.pbn, 1);
    float* pr = fAlloc((size_t)n * ns * c);
    linear(t1, B.p2, pr, n * ns, 3, c, 0);
    float* w0 = fAlloc((size_t)n * ns * c);
    att_w0_kernel<<<CDIV(n * ns, 256), 256, 0, stream>>>(kk, q, pr, idx, n, ns, c, w0);
    bnrelu(w0, n * ns, c, B.wbn1, 1);
    float* w1o = fAlloc((size_t)n * ns * cs);
    linear(w0, B.w1, w1o, n * ns, c, cs, 0);
    bnrelu(w1o, n * ns, cs, B.wbn2, 1);
    float* w2o = fAlloc((size_t)n * ns * cs);
    linear(w1o, B.w2, w2o, n * ns, cs, cs, 0);
    softmax_ns_kernel<<<CDIV((size_t)n * cs, 256), 256, 0, stream>>>(w2o, n, ns, cs);
    float* xv = fAlloc((size_t)n * ns * c);
    att_xv_kernel<<<CDIV(n * ns, 256), 256, 0, stream>>>(v, pr, idx, n, ns, c, xv);
    att_agg_kernel<<<CDIV((size_t)n * c, 256), 256, 0, stream>>>(xv, w2o, n, ns, c, cs, out);
    off = mark;
  };

  auto block = [&](const float* P, float* X, const Blkp& B, int n, int c, int ns) {
    size_t mark = off;
    float* h = fAlloc((size_t)n * c);
    linw(X, B.lin1, nullptr, h, n, c, c, 0);
    bnrelu(h, n, c, B.bn1, 1);
    float* h2 = fAlloc((size_t)n * c);
    pt(P, h, B, n, c, ns, h2);
    bnrelu(h2, n, c, B.bn2, 1);
    float* h3 = fAlloc((size_t)n * c);
    linw(h2, B.lin3, nullptr, h3, n, c, c, 0);
    bnrelu(h3, n, c, B.bn3, 0);
    add_relu_kernel<<<CDIV((size_t)n * c, 256), 256, 0, stream>>>(h3, X, (size_t)n * c);
    off = mark;
  };

  auto tu = [&](const float* P1, const float* X1, int na, const float* P2c,
                const float* X2, int nb, const TUp& T, int cin1, int cin2,
                int cout, float* Y) {
    size_t mark = off;
    linear(X1, T.l1, Y, na, cin1, cout, 0);
    bnrelu(Y, na, cout, T.bn1, 1);
    float* f2 = fAlloc((size_t)nb * cout);
    linear(X2, T.l2, f2, nb, cin2, cout, 0);
    bnrelu(f2, nb, cout, T.bn2, 1);
    int* idx3 = (int*)alloc((size_t)na * 3 * sizeof(int));
    knn_kernel<<<CDIV(na, 64), 64, 0, stream>>>(P1, na, P2c, nb, 3, idx3);
    interp3_kernel<<<CDIV(na, 128), 128, 0, stream>>>(P1, P2c, idx3, f2, Y, na, cout);
    off = mark;
  };

  auto tu_head = [&](const float* X, const TUp& T, int n, int c, float* Y) {
    size_t mark = off;
    float* mu = fAlloc((size_t)c);
    mean_rows_kernel<<<CDIV(c, 256), 256, 0, stream>>>(X, n, c, mu);
    float* gvec = fAlloc((size_t)c);
    linear(mu, T.l2, gvec, 1, c, c, 1);
    float* hcat = fAlloc((size_t)n * 2 * c);
    concat2_kernel<<<CDIV((size_t)n * 2 * c, 256), 256, 0, stream>>>(X, c, gvec, c, n, 1, hcat);
    linear(hcat, T.l1, Y, n, 2 * c, c, 0);
    bnrelu(Y, n, c, T.bn1, 1);
    off = mark;
  };

  // ---- encoder -------------------------------------------------------------
  concat2_kernel<<<CDIV((size_t)n1 * 6, 256), 256, 0, stream>>>(pos, 3, feat, 3, n1, 0, x0);
  {
    size_t mark = off;
    linw(x0, enc[0].w, nullptr, x1, n1, 6, 32, 0);
    BNp bn = {enc[0].b, enc[0].g};
    bnrelu(x1, n1, 32, bn, 1);
    off = mark;
  }
  td(p1, x1, n1, 32,  NS[1], enc[1], 64,  p2, x2);
  td(p2, x2, n2, 64,  NS[2], enc[2], 128, p3, x3);
  td(p3, x3, n3, 128, NS[3], enc[3], 256, p4, x4);
  td(p4, x4, n4, 256, NS[4], enc[4], 512, p5, x5);

  // ---- decoder -------------------------------------------------------------
  tu_head(x5, tud[4], n5, 512, y5);
  block(p5, y5, blkd[4], n5, 512, NS[4]);
  tu(p4, x4, n4, p5, y5, n5, tud[3], 256, 512, 256, y4);
  block(p4, y4, blkd[3], n4, 256, NS[3]);
  tu(p3, x3, n3, p4, y4, n4, tud[2], 128, 256, 128, y3);
  block(p3, y3, blkd[2], n3, 128, NS[2]);
  tu(p2, x2, n2, p3, y3, n3, tud[1], 64, 128, 64, y2);
  block(p2, y2, blkd[1], n2, 64, NS[1]);
  tu(p1, x1, n1, p2, y2, n2, tud[0], 32, 64, 32, y1);
  block(p1, y1, blkd[0], n1, 32, NS[0]);

  // ---- classification head -------------------------------------------------
  {
    size_t mark = off;
    float* hc = fAlloc((size_t)n1 * 32);
    linear(y1, cls_l1, hc, n1, 32, 32, 0);
    bnrelu(hc, n1, 32, cls_bn, 1);
    linear(hc, cls_l2, (float*)d_out, n1, 32, 13, 0);
    off = mark;
  }
}
